// REINFORCE_89842125898146
// MI455X (gfx1250) — matrix operand order repaired and checked by reading, not compiled, not run on hardware
//
#include <hip/hip_runtime.h>

// ---------- types ----------
typedef __attribute__((ext_vector_type(16))) __bf16 v16bf;
typedef __attribute__((ext_vector_type(8)))  float  v8f;
typedef __attribute__((ext_vector_type(4)))  unsigned int v4u;
typedef __attribute__((ext_vector_type(8)))  int    v8i;
typedef __attribute__((ext_vector_type(4)))  int    v4i;

// Native f32->bf16 convert (RNE on gfx1250 hardware)
__device__ __forceinline__ __bf16 f2bf(float f) {
    return (__bf16)f;
}
__device__ __forceinline__ unsigned pack2bf(float a, float b) {
    unsigned short ua = __builtin_bit_cast(unsigned short, (__bf16)a);
    unsigned short ub = __builtin_bit_cast(unsigned short, (__bf16)b);
    return (unsigned)ua | ((unsigned)ub << 16);
}

// ---------- TDM: DMA one 2D bf16 tile (tile_d0 x tile_d1) global -> LDS ----------
// D# per CDNA5 ISA ch.8: group0 = {count/flags, lds_addr, global_addr, type=2},
// group1 = {data_size=2B, tensor dims == tile dims, tile dims, dim0 stride}.
// Rows land contiguously in LDS (row-major tile_d1 x tile_d0).
// This toolchain exposes the 6-arg builtin: (g0, g1, g2, g3, g4, cpol).
__device__ __forceinline__ void tdm_load_tile_2d(unsigned lds_off,
                                                 unsigned long long ga,
                                                 unsigned stride_elems,
                                                 unsigned tile_d0, unsigned tile_d1) {
    v4u g0; v8i g1;
    v4i g2 = {0, 0, 0, 0};
    v4i g3 = {0, 0, 0, 0};
    v8i g4 = {0, 0, 0, 0, 0, 0, 0, 0};
    g0[0] = 1u;                                               // count=1, no gather
    g0[1] = lds_off;                                          // LDS byte address
    g0[2] = (unsigned)ga;                                     // global addr lo
    g0[3] = ((unsigned)(ga >> 32) & 0x01FFFFFFu) | 0x80000000u; // addr hi | type=2
    g1[0] = (int)0x00010000u;                                 // data_size=1 (2 bytes)
    g1[1] = (int)(tile_d0 << 16);                             // tensor_dim0 lo16
    g1[2] = (int)(tile_d1 << 16);                             // td0 hi | tensor_dim1 lo16
    g1[3] = (int)(tile_d0 << 16);                             // td1 hi | tile_dim0
    g1[4] = (int)tile_d1;                                     // tile_dim1 | tile_dim2=0
    g1[5] = (int)stride_elems;                                // tensor_dim0_stride lo32
    g1[6] = 0;                                                // stride hi | dim1_stride lo
    g1[7] = 0;
    __builtin_amdgcn_tensor_load_to_lds(g0, g1, g2, g3, g4, 0);
}

// ---------- weight packing: fp32 row-major [K,N] -> bf16 transposed [N,Kpad] ----------
__global__ __launch_bounds__(256)
void pack_w_t(const float* __restrict__ W, __bf16* __restrict__ Wp,
              int Kreal, int Kpad, int Nout) {
    int i = blockIdx.x * 256 + threadIdx.x;
    if (i >= Nout * Kpad) return;
    int n = i / Kpad;
    int k = i - n * Kpad;
    float v = (k < Kreal) ? W[(size_t)k * Nout + n] : 0.0f;
    Wp[i] = f2bf(v);
}

// ---------- fused MLP GEMM layer ----------
// Out[M,1024] = leakyrelu( A[M,K] @ B[K,1024] + bias ),  bf16 in/out, f32 accumulate.
// MODE 0: A rows gathered on the fly: concat(x_m[mids], x_op[opids], srpt[jobids], pt)
// MODE 1: A is a bf16 buffer [M,1024], staged to LDS via the Tensor Data Mover.
// B (pre-packed transposed weights [1024][K]) is always staged via TDM.
#define BM 64
#define BN 64
#define BK 32

template<int MODE>
__global__ __launch_bounds__(256)
void mlp_gemm(const __bf16* __restrict__ A,
              const float* __restrict__ xm,  const float* __restrict__ xop,
              const float* __restrict__ srpt, const float* __restrict__ ptv,
              const int*   __restrict__ mids, const int* __restrict__ opids,
              const int*   __restrict__ jobids,
              const __bf16* __restrict__ Bp,   // packed weights [1024][K] bf16
              const float* __restrict__ bias,
              __bf16* __restrict__ Out, int M, int K) {
    __shared__ __align__(16) __bf16 As[BM * BK];   // 4 KB
    __shared__ __align__(16) __bf16 Bs[BN * BK];   // 4 KB

    const int t    = threadIdx.x;
    const int lane = t & 31;
    const int wv   = __builtin_amdgcn_readfirstlane(t >> 5);  // wave id, provably scalar
    const int m0   = (wv & 3) * 16;                 // wave's M tile inside 64
    const int nb   = (wv >> 2) * 16;                // wave's first N tile (second at +32)
    const int blockM = blockIdx.x * BM;
    const int blockN = blockIdx.y * BN;

    const unsigned As_off = (unsigned)(size_t)(&As[0]);   // flat LDS addr low 32b = LDS offset
    const unsigned Bs_off = (unsigned)(size_t)(&Bs[0]);

    // gather staging assignment (MODE 0): 64 rows x 32 k, 8 bf16 per thread
    const int arow = t >> 2;            // 0..63
    const int ak0  = (t & 3) * 8;       // 0,8,16,24

    // hoisted per-row gather state (MODE 0 only)
    int   mid = 0, oid = 0;
    float tailA = 0.0f, tailB = 0.0f;
    if (MODE == 0) {
        const int r = blockM + arow;
        mid   = mids[r];
        oid   = opids[r];
        tailA = srpt[jobids[r]];
        tailB = ptv[r];
    }

    v8f acc0 = {}, acc1 = {};

    for (int kt = 0; kt < K; kt += BK) {
        // ---- stage A tile ----
        if (MODE == 1) {
            if (wv == 0) {   // one TDM descriptor moves the whole 64x32 bf16 tile
                tdm_load_tile_2d(As_off,
                                 (unsigned long long)(size_t)A +
                                     ((size_t)blockM * 1024 + kt) * 2,
                                 1024u, BK, BM);
                __builtin_amdgcn_s_wait_tensorcnt(0);
            }
        } else {
            // region is k-tile-uniform (1024/2048 are multiples of BK)
            uint4 p;
            if (kt < 1024) {
                const float* s = xm + (size_t)mid * 1024 + kt + ak0;
                float4 u0 = *(const float4*)(s);
                float4 u1 = *(const float4*)(s + 4);
                p.x = pack2bf(u0.x, u0.y); p.y = pack2bf(u0.z, u0.w);
                p.z = pack2bf(u1.x, u1.y); p.w = pack2bf(u1.z, u1.w);
            } else if (kt < 2048) {
                const float* s = xop + (size_t)oid * 1024 + (kt - 1024) + ak0;
                float4 u0 = *(const float4*)(s);
                float4 u1 = *(const float4*)(s + 4);
                p.x = pack2bf(u0.x, u0.y); p.y = pack2bf(u0.z, u0.w);
                p.z = pack2bf(u1.x, u1.y); p.w = pack2bf(u1.z, u1.w);
            } else {           // boundary tile: k = 2048 (srpt), 2049 (pt), rest zero
                p.x = (ak0 == 0) ? pack2bf(tailA, tailB) : 0u;
                p.y = 0u; p.z = 0u; p.w = 0u;
            }
            *(uint4*)(As + arow * BK + ak0) = p;
        }
        // ---- stage B tile via TDM (64 rows of packed [n][k] weights) ----
        if (wv == 1) {
            tdm_load_tile_2d(Bs_off,
                             (unsigned long long)(size_t)Bp +
                                 ((size_t)blockN * K + kt) * 2,
                             (unsigned)K, BK, BN);
            __builtin_amdgcn_s_wait_tensorcnt(0);
        }

        __syncthreads();

        // ---- per-lane fragments (ISA 7.12.2 16-bit A/B layouts) ----
        const int half = lane >> 4;
        const int r16  = lane & 15;
        union { v16bf v; uint4 q[2]; } fa, fb0, fb1;
        // A 16x32: group0 holds K 0..7 & 16..23, group1 holds K 8..15 & 24..31
        const __bf16* ab = As + (m0 + r16) * BK;
        fa.q[0] = *(const uint4*)(ab + half * 8);
        fa.q[1] = *(const uint4*)(ab + 16 + half * 8);
        // B 32x16: group0 holds K 0..7 & 16..23, group1 holds K 8..15 & 24..31 (column = r16)
        const __bf16* bb0 = Bs + (nb + r16) * BK + half * 8;
        fb0.q[0] = *(const uint4*)(bb0);
        fb0.q[1] = *(const uint4*)(bb0 + 16);
        const __bf16* bb1 = Bs + (nb + 32 + r16) * BK + half * 8;
        fb1.q[0] = *(const uint4*)(bb1);
        fb1.q[1] = *(const uint4*)(bb1 + 16);

        acc0 = __builtin_amdgcn_wmma_f32_16x16x32_bf16(
                   false, fa.v, false, fb0.v, (short)0, acc0, false, false);
        acc1 = __builtin_amdgcn_wmma_f32_16x16x32_bf16(
                   false, fa.v, false, fb1.v, (short)0, acc1, false, false);

        __syncthreads();
    }

    // ---- epilogue: bias + leaky-ReLU, write bf16 ----
    const int half = lane >> 4;
    const int nl   = lane & 15;
    const int c0   = blockN + nb + nl;
    const int c1   = c0 + 32;
    const float bc0 = bias[c0];
    const float bc1 = bias[c1];
    #pragma unroll
    for (int d = 0; d < 8; ++d) {
        const int r = blockM + m0 + d + 8 * half;   // C/D layout: VGPR d -> M=d (+8 lanes>=16)
        float v0 = acc0[d] + bc0;
        float v1 = acc1[d] + bc1;
        v0 = (v0 > 0.0f) ? v0 : 0.01f * v0;
        v1 = (v1 > 0.0f) ? v1 : 0.01f * v1;
        Out[(size_t)r * 1024 + c0] = f2bf(v0);
        Out[(size_t)r * 1024 + c1] = f2bf(v1);
    }
}

// ---------- final projection: score[r] = dot(H3[r,:], W3) + b3 ----------
__global__ __launch_bounds__(256)
void score_kernel(const __bf16* __restrict__ H, const float* __restrict__ W3,
                  const float* __restrict__ b3, float* __restrict__ score, int M) {
    const int row  = blockIdx.x * 8 + (threadIdx.x >> 5);
    const int lane = threadIdx.x & 31;
    if (row >= M) return;
    const __bf16* hp = H + (size_t)row * 1024;
    float s = 0.0f;
    #pragma unroll 4
    for (int k = lane; k < 1024; k += 32) s += (float)hp[k] * W3[k];
    #pragma unroll
    for (int off = 16; off; off >>= 1) s += __shfl_xor(s, off, 32);
    if (lane == 0) score[row] = s + b3[0];
}

// ---------- softmax over N scores + argmax (single block) ----------
__global__ __launch_bounds__(1024)
void softmax_argmax(const float* __restrict__ score, float* __restrict__ out, int N) {
    __shared__ float red[32];
    __shared__ int   redi[32];
    __shared__ float s_gmax, s_gsum;
    __shared__ int   s_gidx;
    const int t = threadIdx.x, lane = t & 31, w = t >> 5;

    float m = -3.402823466e38f; int mi = 0x7fffffff;
    for (int i = t; i < N; i += 1024) { float v = score[i]; if (v > m) { m = v; mi = i; } }
    #pragma unroll
    for (int off = 16; off; off >>= 1) {
        float om = __shfl_xor(m, off, 32);
        int   oi = __shfl_xor(mi, off, 32);
        if (om > m || (om == m && oi < mi)) { m = om; mi = oi; }
    }
    if (lane == 0) { red[w] = m; redi[w] = mi; }
    __syncthreads();
    if (w == 0) {
        float m2 = red[lane]; int i2 = redi[lane];
        #pragma unroll
        for (int off = 16; off; off >>= 1) {
            float om = __shfl_xor(m2, off, 32);
            int   oi = __shfl_xor(i2, off, 32);
            if (om > m2 || (om == m2 && oi < i2)) { m2 = om; i2 = oi; }
        }
        if (lane == 0) { s_gmax = m2; s_gidx = i2; }
    }
    __syncthreads();
    const float gmax = s_gmax;

    float ls = 0.0f;
    for (int i = t; i < N; i += 1024) ls += __expf(score[i] - gmax);
    #pragma unroll
    for (int off = 16; off; off >>= 1) ls += __shfl_xor(ls, off, 32);
    if (lane == 0) red[w] = ls;
    __syncthreads();
    if (w == 0) {
        float s2 = red[lane];
        #pragma unroll
        for (int off = 16; off; off >>= 1) s2 += __shfl_xor(s2, off, 32);
        if (lane == 0) s_gsum = s2;
    }
    __syncthreads();
    const float inv = 1.0f / s_gsum;

    for (int i = t; i < N; i += 1024) out[i] = __expf(score[i] - gmax) * inv;
    if (t == 0) out[N] = (float)s_gidx;
}

// ---------- launch ----------
extern "C" void kernel_launch(void* const* d_in, const int* in_sizes, int n_in,
                              void* d_out, int out_size, void* d_ws, size_t ws_size,
                              hipStream_t stream) {
    const float* x_m    = (const float*)d_in[0];
    const float* x_op   = (const float*)d_in[1];
    const float* srpt   = (const float*)d_in[2];
    const float* pt     = (const float*)d_in[3];
    const float* W0     = (const float*)d_in[4];
    const float* b0     = (const float*)d_in[5];
    const float* W1     = (const float*)d_in[6];
    const float* b1     = (const float*)d_in[7];
    const float* W2     = (const float*)d_in[8];
    const float* b2     = (const float*)d_in[9];
    const float* W3     = (const float*)d_in[10];
    const float* b3     = (const float*)d_in[11];
    const int*   mids   = (const int*)d_in[12];
    const int*   opids  = (const int*)d_in[13];
    const int*   jobids = (const int*)d_in[14];

    const int M     = in_sizes[3];     // 32768 available ops
    const int K0    = 2050;            // 2H+2
    const int K0pad = 2080;            // next multiple of 32

    // workspace layout
    char* ws = (char*)d_ws;
    const size_t OFF_W0P = 0;
    const size_t OFF_W1P = OFF_W0P + (size_t)K0pad * 1024 * 2;
    const size_t OFF_W2P = OFF_W1P + (size_t)1024 * 1024 * 2;
    const size_t OFF_H1  = OFF_W2P + (size_t)1024 * 1024 * 2;
    const size_t szH     = (size_t)M * 1024 * 2;
    const size_t OFF_H2  = OFF_H1 + szH;
    const size_t OFF_SC  = OFF_H2 + szH;

    __bf16* W0p    = (__bf16*)(ws + OFF_W0P);
    __bf16* W1p    = (__bf16*)(ws + OFF_W1P);
    __bf16* W2p    = (__bf16*)(ws + OFF_W2P);
    __bf16* H1     = (__bf16*)(ws + OFF_H1);   // layer1 out; layer3 out reuses this
    __bf16* H2     = (__bf16*)(ws + OFF_H2);
    float*  scorev = (float*)(ws + OFF_SC);
    float*  out    = (float*)d_out;

    // 1) pack weights (fp32 [K,N] -> bf16 [N,Kpad])
    pack_w_t<<<(1024 * K0pad + 255) / 256, 256, 0, stream>>>(W0, W0p, K0, K0pad, 1024);
    pack_w_t<<<(1024 * 1024 + 255) / 256, 256, 0, stream>>>(W1, W1p, 1024, 1024, 1024);
    pack_w_t<<<(1024 * 1024 + 255) / 256, 256, 0, stream>>>(W2, W2p, 1024, 1024, 1024);

    // 2) three WMMA GEMM layers (64x64 block tile, TDM-fed LDS)
    dim3 grid(M / BM, 1024 / BN);
    mlp_gemm<0><<<grid, 256, 0, stream>>>(nullptr, x_m, x_op, srpt, pt,
                                          mids, opids, jobids,
                                          W0p, b0, H1, M, K0pad);
    mlp_gemm<1><<<grid, 256, 0, stream>>>(H1, nullptr, nullptr, nullptr, nullptr,
                                          nullptr, nullptr, nullptr,
                                          W1p, b1, H2, M, 1024);
    mlp_gemm<1><<<grid, 256, 0, stream>>>(H2, nullptr, nullptr, nullptr, nullptr,
                                          nullptr, nullptr, nullptr,
                                          W2p, b2, H1 /*H3 reuses H1*/, M, 1024);

    // 3) final projection + softmax/argmax
    score_kernel<<<M / 8, 256, 0, stream>>>(H1, W3, b3, scorev, M);
    softmax_argmax<<<1, 1024, 0, stream>>>(scorev, out, M);
}
